// GAT_81793357185285
// MI455X (gfx1250) — matrix operand order repaired and checked by reading, not compiled, not run on hardware
//
#include <hip/hip_runtime.h>
#include <math.h>

// ---------------- problem constants (match reference) ----------------
#define NN    20000
#define NE    320000
#define TT    12
#define IND   128
#define HIDD  256
#define OUTD  128
#define NH    4
#define NEG_SLOPE 0.2f

typedef __attribute__((ext_vector_type(16))) __bf16 bf16x16;
typedef __attribute__((ext_vector_type(8)))  float  floatx8;

union Frag { bf16x16 v; float4 q[2]; };

__device__ __forceinline__ unsigned short f2bf(float f) {
    unsigned u = __float_as_uint(f);
    u += 0x7FFFu + ((u >> 16) & 1u);          // round-to-nearest-even
    return (unsigned short)(u >> 16);
}

// ---------------- utility kernels ----------------
__global__ void k_zero_int(int* p, int n) {
    int i = blockIdx.x * blockDim.x + threadIdx.x;
    if (i < n) p[i] = 0;
}

// x[N,T,IND] slice t -> bf16 [N,IND]
__global__ void k_convert_x(const float* __restrict__ x, unsigned short* __restrict__ xb, int t) {
    int i = blockIdx.x * blockDim.x + threadIdx.x;
    if (i >= NN * IND) return;
    int n = i / IND, f = i % IND;
    xb[i] = f2bf(x[((size_t)n * TT + t) * IND + f]);
}

// pack W[K,N] (row-major, fp32) into per-lane WMMA B-fragment layout (bf16)
// layout: [colTile][kk][lane][e], lane: n=colTile*16+(lane&15), k=kk*32+(lane>>4)*16+e
__global__ void k_pack_b(const float* __restrict__ W, unsigned short* __restrict__ Bp, int K, int N) {
    int tid = blockIdx.x * blockDim.x + threadIdx.x;
    if (tid >= K * N) return;
    int KK = K >> 5;
    int e = tid & 15;
    int lane = (tid >> 4) & 31;
    int rest = tid >> 9;
    int kk = rest % KK;
    int colTile = rest / KK;
    int n = colTile * 16 + (lane & 15);
    int k = kk * 32 + (lane >> 4) * 16 + e;
    Bp[tid] = f2bf(W[(size_t)k * N + n]);
}

// ---------------- WMMA GEMM: C[M,N] = A[M,K](bf16) * Bpacked ----------------
// block = 256 threads = 8 waves; wave w -> row tile (blockIdx.x*8+w), 64-col group blockIdx.y
__global__ void k_gemm_wmma(const unsigned short* __restrict__ A,
                            const unsigned short* __restrict__ Bp,
                            float* __restrict__ C, int M, int K, int N) {
    int wave = threadIdx.x >> 5;
    int lane = threadIdx.x & 31;
    int rowTile = blockIdx.x * 8 + wave;
    if (rowTile * 16 >= M) return;                 // wave-uniform exit; EXEC stays all-1s
    int colGroup = blockIdx.y;
    int half   = lane >> 4;
    int mlocal = lane & 15;
    int m = rowTile * 16 + mlocal;
    int KK = K >> 5;

    floatx8 acc[4];
    #pragma unroll
    for (int j = 0; j < 4; ++j) acc[j] = (floatx8){0,0,0,0,0,0,0,0};

    const float4* Arow = reinterpret_cast<const float4*>(A + (size_t)m * K);
    for (int kk = 0; kk < KK; ++kk) {
        Frag a;
        int k0 = kk * 32 + half * 16;              // contiguous A layout: k = half*16 + e (matches packed B)
        a.q[0] = Arow[k0 >> 3];
        a.q[1] = Arow[(k0 + 8) >> 3];
        #pragma unroll
        for (int j = 0; j < 4; ++j) {
            Frag b;
            const float4* bp = reinterpret_cast<const float4*>(
                Bp + ((((size_t)(colGroup * 4 + j) * KK + kk) * 32 + lane) << 4));
            b.q[0] = bp[0];
            b.q[1] = bp[1];
            acc[j] = __builtin_amdgcn_wmma_f32_16x16x32_bf16(
                false, a.v, false, b.v, (short)0, acc[j], false, false);
        }
    }
    #pragma unroll
    for (int j = 0; j < 4; ++j) {
        int col = colGroup * 64 + j * 16 + mlocal;
        #pragma unroll
        for (int r = 0; r < 8; ++r) {
            int row = rowTile * 16 + half * 8 + r; // C/D layout: m = r + half*8
            C[(size_t)row * N + col] = acc[j][r];
        }
    }
}

// ---------------- CSR build (once per launch; graph is static) ----------------
__global__ void k_hist(const long long* __restrict__ dst, int* __restrict__ deg) {
    int e = blockIdx.x * blockDim.x + threadIdx.x;
    if (e < NE) atomicAdd(&deg[(int)dst[e]], 1);
}

// single-block exclusive scan of deg[NN] -> rowptr[NN+1], cursor[NN]
__global__ void k_scan(const int* __restrict__ deg, int* __restrict__ rowptr,
                       int* __restrict__ cursor) {
    __shared__ int sdata[1024];
    const int T_ = 1024, CH = (NN + T_ - 1) / T_;   // 20 per thread
    int tid = threadIdx.x;
    int base = tid * CH;
    int s = 0;
    for (int j = 0; j < CH; ++j) { int idx = base + j; if (idx < NN) s += deg[idx]; }
    sdata[tid] = s;
    __syncthreads();
    for (int off = 1; off < T_; off <<= 1) {        // inclusive Hillis-Steele scan
        int v = sdata[tid];
        int u = (tid >= off) ? sdata[tid - off] : 0;
        __syncthreads();
        sdata[tid] = v + u;
        __syncthreads();
    }
    int run = sdata[tid] - s;                       // exclusive chunk base
    for (int j = 0; j < CH; ++j) {
        int idx = base + j;
        if (idx < NN) { rowptr[idx] = run; cursor[idx] = run; run += deg[idx]; }
    }
    if (tid == T_ - 1) rowptr[NN] = run;
}

__global__ void k_place(const long long* __restrict__ src, const long long* __restrict__ dst,
                        int* __restrict__ cursor, int* __restrict__ esrc) {
    int e = blockIdx.x * blockDim.x + threadIdx.x;
    if (e >= NE) return;
    int pos = atomicAdd(&cursor[(int)dst[e]], 1);
    esrc[pos] = (int)src[e];
}

// sort each node's bucket by src (duplicate (src,dst) edges are numerically
// identical, so this makes the per-node reduction order deterministic)
__global__ void k_sort(const int* __restrict__ rowptr, int* __restrict__ esrc) {
    int n = blockIdx.x * blockDim.x + threadIdx.x;
    if (n >= NN) return;
    int b = rowptr[n], e = rowptr[n + 1];
    for (int i = b + 1; i < e; ++i) {
        int v = esrc[i], j = i - 1;
        while (j >= b && esrc[j] > v) { esrc[j + 1] = esrc[j]; --j; }
        esrc[j + 1] = v;
    }
}

// ---------------- attention coefficients ----------------
__global__ void k_alpha(const float* __restrict__ h, const float* __restrict__ a_src,
                        const float* __restrict__ a_dst, float* __restrict__ as,
                        float* __restrict__ ad, int Nout, int C) {
    int i = blockIdx.x * blockDim.x + threadIdx.x;
    if (i >= NN * NH) return;
    int n = i / NH, hd = i % NH;
    const float* hp = h + (size_t)n * Nout + hd * C;
    const float* s  = a_src + hd * C;
    const float* d  = a_dst + hd * C;
    float ss = 0.f, dd = 0.f;
    for (int c = 0; c < C; ++c) { float v = hp[c]; ss += v * s[c]; dd += v * d[c]; }
    as[i] = ss; ad[i] = dd;
}

// per-(node,head) gather softmax over in-edges; writes normalized alpha. No atomics.
__global__ void k_attn(const int* __restrict__ rowptr, const int* __restrict__ esrc,
                       const float* __restrict__ as, const float* __restrict__ ad,
                       float* __restrict__ wbuf) {
    int i = blockIdx.x * blockDim.x + threadIdx.x;
    if (i >= NN * NH) return;
    int n = i / NH, hd = i % NH;
    int b = rowptr[n], e = rowptr[n + 1];
    if (b == e) return;
    float adv = ad[n * NH + hd];
    float m = -INFINITY;
    for (int p = b; p < e; ++p) {
        float v = as[esrc[p] * NH + hd] + adv;
        v = (v > 0.f) ? v : NEG_SLOPE * v;
        m = fmaxf(m, v);
    }
    float denom = 0.f;
    for (int p = b; p < e; ++p) {
        float v = as[esrc[p] * NH + hd] + adv;
        v = (v > 0.f) ? v : NEG_SLOPE * v;
        float w = expf(v - m);
        wbuf[p * NH + hd] = w;
        denom += w;
    }
    float sc = 1.0f / (denom + 1e-16f);
    for (int p = b; p < e; ++p) wbuf[p * NH + hd] *= sc;
}

// gather-aggregate + bias (+relu) + bf16 for next layer (+ optional fp32 output)
// one block per node, one thread per output channel; h-row reads fully coalesced
__global__ void k_aggr(const int* __restrict__ rowptr, const int* __restrict__ esrc,
                       const float* __restrict__ wbuf, const float* __restrict__ h,
                       const float* __restrict__ bias, unsigned short* __restrict__ xb_next,
                       float* __restrict__ out, int t, int Nout, int C, int doRelu) {
    int n = blockIdx.x;
    int c = threadIdx.x;                 // blockDim.x == Nout
    int hd = c / C;
    int b = rowptr[n], e = rowptr[n + 1];
    float acc = 0.f;
    for (int p = b; p < e; ++p)
        acc += wbuf[p * NH + hd] * h[(size_t)esrc[p] * Nout + c];
    float v = acc + bias[c];
    if (doRelu) v = fmaxf(v, 0.f);
    xb_next[(size_t)n * Nout + c] = f2bf(v);
    if (out) out[((size_t)n * TT + t) * OUTD + c] = v;
}

// ---------------- host-side launcher ----------------
static inline int cdiv(int a, int b) { return (a + b - 1) / b; }

extern "C" void kernel_launch(void* const* d_in, const int* in_sizes, int n_in,
                              void* d_out, int out_size, void* d_ws, size_t ws_size,
                              hipStream_t stream) {
    const float*     x    = (const float*)d_in[0];
    const long long* ei   = (const long long*)d_in[1];
    const long long* srcI = ei;
    const long long* dstI = ei + NE;
    const float* W[3]     = {(const float*)d_in[2], (const float*)d_in[6], (const float*)d_in[10]};
    const float* aS[3]    = {(const float*)d_in[3], (const float*)d_in[7], (const float*)d_in[11]};
    const float* aD[3]    = {(const float*)d_in[4], (const float*)d_in[8], (const float*)d_in[12]};
    const float* Bi[3]    = {(const float*)d_in[5], (const float*)d_in[9], (const float*)d_in[13]};
    float* out = (float*)d_out;

    // workspace carve-out (256-B aligned)
    char* ws = (char*)d_ws;
    size_t off = 0;
    auto carve = [&](size_t bytes) { char* p = ws + off; off = (off + bytes + 255) & ~(size_t)255; return p; };
    unsigned short* xb     = (unsigned short*)carve((size_t)NN * HIDD * 2);
    float*          hmat   = (float*)carve((size_t)NN * HIDD * 4);
    float*          as     = (float*)carve((size_t)NN * NH * 4);
    float*          ad     = (float*)carve((size_t)NN * NH * 4);
    float*          wbuf   = (float*)carve((size_t)NE * NH * 4);
    int*            deg    = (int*)carve((size_t)NN * 4);
    int*            rowptr = (int*)carve((size_t)(NN + 1) * 4);
    int*            cursor = (int*)carve((size_t)NN * 4);
    int*            esrc   = (int*)carve((size_t)NE * 4);
    unsigned short* Wp[3];
    Wp[0] = (unsigned short*)carve((size_t)IND  * HIDD * 2);
    Wp[1] = (unsigned short*)carve((size_t)HIDD * HIDD * 2);
    Wp[2] = (unsigned short*)carve((size_t)HIDD * OUTD * 2);

    const int BLK = 256;
    const int Kin[3]  = {IND, HIDD, HIDD};
    const int Nout[3] = {HIDD, HIDD, OUTD};
    const int Cdim[3] = {HIDD / NH, HIDD / NH, OUTD / NH};

    // ---- build CSR-by-dst once per launch (graph is static across t/layers) ----
    k_zero_int<<<cdiv(NN, BLK), BLK, 0, stream>>>(deg, NN);
    k_hist    <<<cdiv(NE, BLK), BLK, 0, stream>>>(dstI, deg);
    k_scan    <<<1, 1024, 0, stream>>>(deg, rowptr, cursor);
    k_place   <<<cdiv(NE, BLK), BLK, 0, stream>>>(srcI, dstI, cursor, esrc);
    k_sort    <<<cdiv(NN, BLK), BLK, 0, stream>>>(rowptr, esrc);

    // pack weights into WMMA B-fragment layout (once per launch)
    for (int l = 0; l < 3; ++l)
        k_pack_b<<<cdiv(Kin[l] * Nout[l], BLK), BLK, 0, stream>>>(W[l], Wp[l], Kin[l], Nout[l]);

    const int rowTiles = NN / 16;   // 1250
    for (int t = 0; t < TT; ++t) {
        k_convert_x<<<cdiv(NN * IND, BLK), BLK, 0, stream>>>(x, xb, t);
        for (int l = 0; l < 3; ++l) {
            const int K = Kin[l], N = Nout[l], C = Cdim[l];
            // h = xb @ W  (bf16 WMMA, fp32 accumulate)
            dim3 ggrid(cdiv(rowTiles, 8), N / 64);
            k_gemm_wmma<<<ggrid, BLK, 0, stream>>>(xb, Wp[l], hmat, NN, K, N);
            // attention coefficients
            k_alpha<<<cdiv(NN * NH, BLK), BLK, 0, stream>>>(hmat, aS[l], aD[l], as, ad, N, C);
            // gather softmax over in-edges (atomic-free)
            k_attn<<<cdiv(NN * NH, BLK), BLK, 0, stream>>>(rowptr, esrc, as, ad, wbuf);
            // gather aggregation + bias (+relu) + bf16 next-layer input; layer 2 writes d_out
            k_aggr<<<NN, N, 0, stream>>>(rowptr, esrc, wbuf, hmat, Bi[l], xb,
                                         (l == 2) ? out : nullptr, t, N, C, (l < 2) ? 1 : 0);
        }
    }
    (void)in_sizes; (void)n_in; (void)out_size; (void)ws_size;
}
